// GroupedQueryAttention_64287070486905
// MI455X (gfx1250) — matrix operand order repaired and checked by reading, not compiled, not run on hardware
//
#include <hip/hip_runtime.h>
#include <hip/hip_bf16.h>

// ---------------- static problem shape (from reference) ----------------
#define B_     8
#define S_     16
#define E_     4096
#define HQ_    32
#define HKV_   8
#define HD_    128
#define RATIO_ 4
#define QKVN_  (E_ + 2 * HKV_ * HD_)   // 6144
#define M_     (B_ * S_)               // 128 token rows

typedef __attribute__((ext_vector_type(16))) __bf16 v16bf;
typedef __attribute__((ext_vector_type(8)))  float  v8f;
typedef int v4i __attribute__((vector_size(16)));   // matches builtin param type

#define NEG_BIG (-1.0e30f)

// -------- CDNA5 async global->LDS staging (ASYNCcnt path), guarded --------
#if defined(__HIP_DEVICE_COMPILE__) && __has_builtin(__builtin_amdgcn_global_load_async_to_lds_b128)
#define HAVE_ASYNC_LDS 1
static __device__ __forceinline__ void async_g2l_b128(const float* g, float* l) {
  __builtin_amdgcn_global_load_async_to_lds_b128(
      (__attribute__((address_space(1))) v4i*)g,
      (__attribute__((address_space(3))) v4i*)l,
      /*offset=*/0, /*cpol=*/0);
}
static __device__ __forceinline__ void async_wait0() {
#if __has_builtin(__builtin_amdgcn_s_wait_asynccnt)
  __builtin_amdgcn_s_wait_asynccnt(0);
#else
  asm volatile("s_wait_asynccnt 0x0" ::: "memory");
#endif
}
#else
#define HAVE_ASYNC_LDS 0
#endif

static __device__ __forceinline__ v8f wmma_bf16(v16bf a, v16bf b, v8f c) {
  // D = A(16x32) * B(32x16) + C, f32 accumulate
  return __builtin_amdgcn_wmma_f32_16x16x32_bf16(
      /*neg_a=*/false, a, /*neg_b=*/false, b,
      /*c_mod=*/(short)0, c, /*reuse_a=*/false, /*reuse_b=*/false);
}

// batched fp32->bf16 fragment convert (4x float4 -> 16 bf16 lanes-elements)
static __device__ __forceinline__ v16bf cvt16(float4 a, float4 b, float4 c, float4 d) {
  v16bf r;
  r[0]  = (__bf16)a.x; r[1]  = (__bf16)a.y; r[2]  = (__bf16)a.z; r[3]  = (__bf16)a.w;
  r[4]  = (__bf16)b.x; r[5]  = (__bf16)b.y; r[6]  = (__bf16)b.z; r[7]  = (__bf16)b.w;
  r[8]  = (__bf16)c.x; r[9]  = (__bf16)c.y; r[10] = (__bf16)c.z; r[11] = (__bf16)c.w;
  r[12] = (__bf16)d.x; r[13] = (__bf16)d.y; r[14] = (__bf16)d.z; r[15] = (__bf16)d.w;
  return r;
}
static __device__ __forceinline__ v16bf frag_from_lds16(const float* p) {
  // 16 contiguous fp32 -> one bf16 fragment; 4 ds_load_b128 issued together
  float4 a = *(const float4*)(p + 0);
  float4 b = *(const float4*)(p + 4);
  float4 c = *(const float4*)(p + 8);
  float4 d = *(const float4*)(p + 12);
  return cvt16(a, b, c, d);
}
static __device__ __forceinline__ v16bf frag_from_lds8x2(const float* p0, const float* p1) {
  float4 a = *(const float4*)(p0 + 0);
  float4 b = *(const float4*)(p0 + 4);
  float4 c = *(const float4*)(p1 + 0);
  float4 d = *(const float4*)(p1 + 4);
  return cvt16(a, b, c, d);
}

// =====================================================================
// GEMM: Out[128,N] = A[128,K] @ W[N,K]^T   (fp32 in/out, bf16 WMMA core)
// Block: 256 threads (8 waves). Each wave: 16 M-rows x 128 N-cols tile.
// =====================================================================
__global__ __launch_bounds__(256) void gemm128_bf16(
    const float* __restrict__ A,
    const float* __restrict__ W,
    float* __restrict__ Out,
    int N, int K)
{
  __shared__ __align__(16) float As[128 * 36];  // padded stride 36
  __shared__ __align__(16) float Ws[128 * 36];

  const int n0    = blockIdx.x * 128;
  const int tid   = threadIdx.x;
  const int wv    = tid >> 5;
  const int lane  = tid & 31;
  const int m0    = wv * 16;
  const int lhalf = lane >> 4;
  const int l16   = lane & 15;

  v8f acc[8];
  for (int j = 0; j < 8; j++)
    for (int i = 0; i < 8; i++) acc[j][i] = 0.0f;

  const int lr = tid >> 3;         // 0..31  (row base for coop loads)
  const int lc = (tid & 7) * 4;    // 0..28  (float4 column)

  for (int k0 = 0; k0 < K; k0 += 32) {
    __syncthreads();
#if HAVE_ASYNC_LDS
    // async DMA the 128x32 fp32 tiles straight into LDS (no VGPR round trip)
    for (int rr = 0; rr < 128; rr += 32) {
      const int r = lr + rr;
      async_g2l_b128(A + (size_t)r * K + k0 + lc,        &As[r * 36 + lc]);
      async_g2l_b128(W + (size_t)(n0 + r) * K + k0 + lc, &Ws[r * 36 + lc]);
    }
    async_wait0();
#else
    for (int rr = 0; rr < 128; rr += 32) {
      const int r = lr + rr;
      float4 va = *(const float4*)(A + (size_t)r * K + k0 + lc);
      float4 vw = *(const float4*)(W + (size_t)(n0 + r) * K + k0 + lc);
      *(float4*)(&As[r * 36 + lc]) = va;
      *(float4*)(&Ws[r * 36 + lc]) = vw;
    }
    if (k0 + 32 < K) {  // hint next K tile -> global_prefetch
      __builtin_prefetch(A + (size_t)lr * K + k0 + 32 + lc, 0, 1);
      __builtin_prefetch(W + (size_t)(n0 + lr) * K + k0 + 32 + lc, 0, 1);
    }
#endif
    __syncthreads();

    // A fragment (16x32 bf16), ISA 7.12.2 layout:
    // lanes 0-15: row M=lane, K {0..7,16..23}; lanes 16-31: K {8..15,24..31}
    const int kh = lhalf ? 8 : 0;
    const int arow = m0 + l16;
    v16bf af = frag_from_lds8x2(&As[arow * 36 + kh], &As[arow * 36 + 16 + kh]);

    // B fragments: lane%16 = N col, K order same as A: {0..7,16..23}/{8..15,24..31}
    const int kb = lhalf ? 16 : 0;
#pragma unroll
    for (int j = 0; j < 8; j++) {
      const int wrow = 16 * j + l16;
      v16bf bf_ = frag_from_lds8x2(&Ws[wrow * 36 + kh], &Ws[wrow * 36 + 16 + kh]);
      acc[j] = wmma_bf16(af, bf_, acc[j]);
    }
  }

  // C layout: VGPR i, lane -> row = i + 8*(lane/16), col = lane%16
#pragma unroll
  for (int j = 0; j < 8; j++) {
    const int col = n0 + 16 * j + l16;
#pragma unroll
    for (int i = 0; i < 8; i++) {
      const int row = m0 + i + 8 * lhalf;
      Out[(size_t)row * N + col] = acc[j][i];
    }
  }
}

// =====================================================================
// RoPE: rotate q (32 heads) and k (8 heads) slices of qkv.
// =====================================================================
__global__ void rope_kernel(const float* __restrict__ qkv,
                            float* __restrict__ q_rope,
                            float* __restrict__ k_new,
                            const int* __restrict__ offset_p)
{
  const int HALF = HD_ / 2;
  const int total = B_ * S_ * (HQ_ + HKV_) * HALF;
  int idx = blockIdx.x * blockDim.x + threadIdx.x;
  if (idx >= total) return;

  const int j = idx % HALF;
  const int h = (idx / HALF) % (HQ_ + HKV_);
  const int s = (idx / (HALF * (HQ_ + HKV_))) % S_;
  const int b = idx / (HALF * (HQ_ + HKV_) * S_);
  const int offset = *offset_p;

  const float LOG_BASE = 9.210340371976184f;  // ln(10000)
  float inv_freq = __expf(-((float)(2 * j) / (float)HD_) * LOG_BASE);
  float ang = (float)(offset + s) * inv_freq;
  float sn, cs;
  __sincosf(ang, &sn, &cs);

  const float* row = qkv + (size_t)(b * S_ + s) * QKVN_;
  if (h < HQ_) {
    float t1 = row[h * HD_ + j];
    float t2 = row[h * HD_ + j + HALF];
    float* o = q_rope + ((size_t)(b * S_ + s) * HQ_ + h) * HD_;
    o[j]        = t1 * cs - t2 * sn;
    o[j + HALF] = t2 * cs + t1 * sn;
  } else {
    const int hk = h - HQ_;
    float t1 = row[E_ + hk * HD_ + j];
    float t2 = row[E_ + hk * HD_ + j + HALF];
    float* o = k_new + ((size_t)(b * S_ + s) * HKV_ + hk) * HD_;
    o[j]        = t1 * cs - t2 * sn;
    o[j + HALF] = t2 * cs + t1 * sn;
  }
}

// =====================================================================
// Flash attention, one block per (b, kv-head g). 128 threads = 4 waves;
// wave r handles q-head h = g*RATIO + r (16 query rows = S).
// K tiles: async global->LDS DMA when fully inside the cache.
// V tiles: staged transposed (Vst[d][key]) so P*V B-fragments are
// contiguous float4 LDS reads.
// =====================================================================
__global__ __launch_bounds__(128) void attn_kernel(
    const float* __restrict__ q_rope,   // [B,S,HQ,HD]
    const float* __restrict__ k_cache,  // [B,Tc,HKV,HD]
    const float* __restrict__ v_cache,  // [B,Tc,HKV,HD]
    const float* __restrict__ k_new,    // [B,S,HKV,HD]
    const float* __restrict__ qkv,      // v_new lives in qkv slice
    float* __restrict__ attn_out,       // [B,S,E]
    const int* __restrict__ offset_p,
    int Tc)
{
  __shared__ __align__(16) float Ks[32 * 132];    // [key][d], padded
  __shared__ __align__(16) float Vst[128 * 36];   // [d][key], padded (transposed)
  __shared__ __align__(16) float Ps[4][16 * 36];  // per-wave P relayout pad

  const int b    = blockIdx.x / HKV_;
  const int g    = blockIdx.x % HKV_;
  const int tid  = threadIdx.x;
  const int wv   = tid >> 5;
  const int lane = tid & 31;
  const int lhalf= lane >> 4;
  const int l16  = lane & 15;
  const int kb   = lhalf ? 16 : 0;
  const int kh   = lhalf ? 8 : 0;

  const int offset = *offset_p;
  const int T = offset + S_;
  const bool causal = (offset == 0);
  const float scale = rsqrtf((float)HD_);
  const int h = g * RATIO_ + wv;   // query head for this wave

  // Q fragments for 4 K-chunks of HD (pre-scaled by 1/sqrt(HD))
  v16bf aq[4];
  {
    const float* qrow = q_rope + ((size_t)(b * S_ + l16) * HQ_ + h) * HD_;
#pragma unroll
    for (int kc = 0; kc < 4; kc++) {
#pragma unroll
      for (int v = 0; v < 4; v++) {
        const int d = kc * 32 + kh + 2 * v;
        aq[kc][2 * v + 0]     = (__bf16)(qrow[d + 0] * scale);
        aq[kc][2 * v + 1]     = (__bf16)(qrow[d + 1] * scale);
        aq[kc][8 + 2 * v + 0] = (__bf16)(qrow[16 + d + 0] * scale);
        aq[kc][8 + 2 * v + 1] = (__bf16)(qrow[16 + d + 1] * scale);
      }
    }
  }

  v8f O[8];
  for (int j = 0; j < 8; j++)
    for (int i = 0; i < 8; i++) O[j][i] = 0.0f;
  float m_run[8], l_run[8];
  for (int i = 0; i < 8; i++) { m_run[i] = NEG_BIG; l_run[i] = 0.0f; }

  for (int t0 = 0; t0 < T; t0 += 32) {
    __syncthreads();
    {
      const int kr    = tid >> 2;        // 0..31 local key
      const int cbase = (tid & 3) * 32;  // 0/32/64/96
      const int t = t0 + kr;
      const bool tile_in_cache = (t0 + 32 <= offset);  // block-uniform

      // ---- K tile: row-major [key][d] ----
#if HAVE_ASYNC_LDS
      if (tile_in_cache) {
        const size_t base = (((size_t)b * Tc + t) * HKV_ + g) * HD_;
        for (int c4 = 0; c4 < 32; c4 += 4) {
          const int d = cbase + c4;
          async_g2l_b128(k_cache + base + d, &Ks[kr * 132 + d]);
        }
      } else
#endif
      {
        for (int c4 = 0; c4 < 32; c4 += 4) {
          const int d = cbase + c4;
          float4 kv4;
          if (t < T) {
            if (t < offset) {
              kv4 = *(const float4*)(k_cache + (((size_t)b * Tc + t) * HKV_ + g) * HD_ + d);
            } else {
              const int s = t - offset;
              kv4 = *(const float4*)(k_new + ((size_t)(b * S_ + s) * HKV_ + g) * HD_ + d);
            }
          } else {
            kv4 = make_float4(0.f, 0.f, 0.f, 0.f);
          }
          *(float4*)(&Ks[kr * 132 + d]) = kv4;
        }
      }

      // ---- V tile: transposed store Vst[d][key] (manual; DMA can't transpose) ----
      for (int c4 = 0; c4 < 32; c4 += 4) {
        const int d = cbase + c4;
        float4 vv4;
        if (t < T) {
          if (t < offset) {
            vv4 = *(const float4*)(v_cache + (((size_t)b * Tc + t) * HKV_ + g) * HD_ + d);
          } else {
            const int s = t - offset;
            vv4 = *(const float4*)(qkv + (size_t)(b * S_ + s) * QKVN_
                                   + E_ + HKV_ * HD_ + g * HD_ + d);
          }
        } else {
          vv4 = make_float4(0.f, 0.f, 0.f, 0.f);
        }
        Vst[(d + 0) * 36 + kr] = vv4.x;
        Vst[(d + 1) * 36 + kr] = vv4.y;
        Vst[(d + 2) * 36 + kr] = vv4.z;
        Vst[(d + 3) * 36 + kr] = vv4.w;
      }
    }
#if HAVE_ASYNC_LDS
    async_wait0();
#endif
    __syncthreads();

    // scores: c0 = keys t0..t0+15, c1 = keys t0+16..t0+31
    v8f c0, c1;
    for (int i = 0; i < 8; i++) { c0[i] = 0.0f; c1[i] = 0.0f; }
#pragma unroll
    for (int kc = 0; kc < 4; kc++) {
      v16bf b0 = frag_from_lds8x2(&Ks[l16 * 132 + kc * 32 + kh], &Ks[l16 * 132 + kc * 32 + 16 + kh]);
      v16bf b1 = frag_from_lds8x2(&Ks[(l16 + 16) * 132 + kc * 32 + kh], &Ks[(l16 + 16) * 132 + kc * 32 + 16 + kh]);
      c0 = wmma_bf16(aq[kc], b0, c0);
      c1 = wmma_bf16(aq[kc], b1, c1);
    }

    // mask out-of-range / causal, then online softmax update
#pragma unroll
    for (int i = 0; i < 8; i++) {
      const int s  = i + 8 * lhalf;
      const int t_a = t0 + l16;
      const int t_b = t_a + 16;
      if (t_a >= T || (causal && t_a > s)) c0[i] = NEG_BIG;
      if (t_b >= T || (causal && t_b > s)) c1[i] = NEG_BIG;
    }
#pragma unroll
    for (int i = 0; i < 8; i++) {
      const int s = i + 8 * lhalf;
      float mt = fmaxf(c0[i], c1[i]);
      for (int mk = 1; mk < 16; mk <<= 1)     // reduce within 16-lane half
        mt = fmaxf(mt, __shfl_xor(mt, mk, 32));
      const float m_new = fmaxf(m_run[i], mt);
      const float alpha = __expf(m_run[i] - m_new);
      const float p0 = __expf(c0[i] - m_new);
      const float p1 = __expf(c1[i] - m_new);
      float rs = p0 + p1;
      for (int mk = 1; mk < 16; mk <<= 1)
        rs += __shfl_xor(rs, mk, 32);
      l_run[i] = l_run[i] * alpha + rs;
      m_run[i] = m_new;
#pragma unroll
      for (int j = 0; j < 8; j++) O[j][i] *= alpha;
      Ps[wv][s * 36 + l16]      = p0;   // C-layout -> LDS for A-relayout
      Ps[wv][s * 36 + 16 + l16] = p1;
    }
    __syncthreads();

    // P fragment (16 rows x 32 keys) in A layout
    v16bf ap = frag_from_lds8x2(&Ps[wv][l16 * 36 + kh], &Ps[wv][l16 * 36 + 16 + kh]);

    // O[16x128] += P[16x32] * V[32x128]; V B-frags contiguous in Vst
#pragma unroll
    for (int j = 0; j < 8; j++) {
      v16bf bv = frag_from_lds8x2(&Vst[(16 * j + l16) * 36 + kh], &Vst[(16 * j + l16) * 36 + 16 + kh]);
      O[j] = wmma_bf16(ap, bv, O[j]);
    }
  }

  // finalize: O / l, scatter to [B,S,E]
#pragma unroll
  for (int i = 0; i < 8; i++) {
    const int s = i + 8 * lhalf;
    const float inv_l = (l_run[i] > 0.0f) ? 1.0f / l_run[i] : 0.0f;
#pragma unroll
    for (int j = 0; j < 8; j++) {
      const int d = 16 * j + l16;
      attn_out[(size_t)(b * S_ + s) * E_ + h * HD_ + d] = O[j][i] * inv_l;
    }
  }
}

// =====================================================================
// launcher
// =====================================================================
extern "C" void kernel_launch(void* const* d_in, const int* in_sizes, int n_in,
                              void* d_out, int out_size, void* d_ws, size_t ws_size,
                              hipStream_t stream) {
  const float* x       = (const float*)d_in[0];
  const float* w_in    = (const float*)d_in[1];
  const float* w_out   = (const float*)d_in[2];
  const float* k_cache = (const float*)d_in[3];
  const float* v_cache = (const float*)d_in[4];
  const int*   offset  = (const int*)d_in[5];
  const int Tc = in_sizes[3] / (B_ * HKV_ * HD_);   // cache length (4096)

  float* ws     = (float*)d_ws;
  float* qkv    = ws;                                 // 128*6144
  float* q_rope = qkv    + (size_t)M_ * QKVN_;        // 128*4096
  float* k_new  = q_rope + (size_t)M_ * E_;           // 128*1024
  float* attn   = k_new  + (size_t)M_ * HKV_ * HD_;   // 128*4096

  // 1) qkv = x @ w_in^T
  gemm128_bf16<<<dim3(QKVN_ / 128), 256, 0, stream>>>(x, w_in, qkv, QKVN_, E_);

  // 2) RoPE on q and k
  {
    const int total = B_ * S_ * (HQ_ + HKV_) * (HD_ / 2);
    rope_kernel<<<(total + 255) / 256, 256, 0, stream>>>(qkv, q_rope, k_new, offset);
  }

  // 3) flash attention (GQA), splicing new K/V with cache at t==offset
  attn_kernel<<<dim3(B_ * HKV_), 128, 0, stream>>>(
      q_rope, k_cache, v_cache, k_new, qkv, attn, offset, Tc);

  // 4) out = attn @ w_out^T
  gemm128_bf16<<<dim3(E_ / 128), 256, 0, stream>>>(attn, w_out, (float*)d_out, E_, E_);
}
